// Arch24GraphEncoder_69329362092550
// MI455X (gfx1250) — matrix operand order repaired and checked by reading, not compiled, not run on hardware
//
#include <hip/hip_runtime.h>
#include <hip/hip_bf16.h>

// ---------------- problem constants (from reference) ----------------
constexpr int HD     = 128;
constexpr int KSUB   = 12;        // nodes per subgraph
constexpr int MSUB   = 4;         // subgraphs per canonical node
constexpr int NTOT   = 2500;
constexpr int SSUB   = 10000;     // NTOT*MSUB
constexpr int FF     = 120000;    // SSUB*KSUB
constexpr int EINTRA = 240000;
constexpr int EINTER = 40000;
constexpr int BB     = 50;
constexpr int LL     = 4;

typedef __attribute__((ext_vector_type(16))) __bf16 v16bf;
typedef __attribute__((ext_vector_type(8)))  float  v8f;

static __device__ __forceinline__ __bf16 f2bf(float f) {
  unsigned u = __builtin_bit_cast(unsigned, f);
  unsigned short s = (unsigned short)(u >> 16);   // truncate-to-bf16
  return __builtin_bit_cast(__bf16, s);
}

// =====================================================================
// Generic 128-wide WMMA GEMM:  Out[r,:] = act(A[map(r),:] @ W + bias (+Cadd[r,:]))
//  - W is [128,128] row-major (k-major), converted to bf16 in LDS transposed.
//  - block = 256 threads = 8 waves; each wave owns 16 rows x 128 cols.
//  - mapRoot: A row index = (r/12)*12  (root broadcast gather)
// =====================================================================
__global__ __launch_bounds__(256) void gemm128_wmma(
    const float* __restrict__ A, const float* __restrict__ W,
    const float* __restrict__ bias, const float* __restrict__ Cadd,
    float* __restrict__ Out, int Mrows, int mapRoot, int reluOut)
{
  __shared__ __bf16 Wl[HD][HD + 8];            // [n][k], padded vs bank conflicts
  const int tid = threadIdx.x;
  for (int i = tid; i < HD * HD; i += 256) {
    int k = i >> 7, n = i & 127;
    Wl[n][k] = f2bf(W[i]);
  }
  __syncthreads();

  const int wave = tid >> 5, lane = tid & 31;
  const int half = lane >> 4, lid = lane & 15;  // lane half selects K sub-range
  const int rowBase = blockIdx.x * 128 + wave * 16;

  int ar = rowBase + lid;                       // A-fragment row for this lane
  if (ar > Mrows - 1) ar = Mrows - 1;           // clamp OOB loads (stores guarded)
  if (mapRoot) ar = (ar / KSUB) * KSUB;
  const float* Arow = A + (size_t)ar * HD;

  v8f acc[8];
#pragma unroll
  for (int t = 0; t < 8; ++t)
#pragma unroll
    for (int j = 0; j < 8; ++j) acc[t][j] = 0.f;

  for (int kk = 0; kk < HD; kk += 32) {
    // A 16x32 bf16 fragment: lanes 0-15 K=kk..kk+15,
    // lanes 16-31 K=kk+16..kk+31, row M = lane%16.
    const int k0 = kk + half * 16;
    v16bf af;
#pragma unroll
    for (int j = 0; j < 16; ++j) { af[j] = f2bf(Arow[k0 + j]); }

    // B 32x16 bf16 fragment: lane holds one column n, lanes 0-15 K=kk..kk+15,
    // lanes 16-31 K=kk+16..kk+31 (contiguous in LDS -> vector ds loads).
    const int kb = kk + half * 16;
#pragma unroll
    for (int nt = 0; nt < 8; ++nt) {
      const __bf16* wp = &Wl[nt * 16 + lid][kb];
      v16bf bf;
#pragma unroll
      for (int j = 0; j < 16; ++j) bf[j] = wp[j];
      acc[nt] = __builtin_amdgcn_wmma_f32_16x16x32_bf16(
          false, af, false, bf, (short)0, acc[nt], false, false);
    }
  }

  // C/D layout: lanes 0-15: VGPR j -> M=j; lanes 16-31: VGPR j -> M=j+8; N=lane%16
#pragma unroll
  for (int nt = 0; nt < 8; ++nt) {
    const int c = nt * 16 + lid;
#pragma unroll
    for (int j = 0; j < 8; ++j) {
      int r = rowBase + half * 8 + j;
      if (r < Mrows) {
        float v = acc[nt][j] + bias[c];
        if (Cadd) v += Cadd[(size_t)r * HD + c];
        if (reluOut) v = fmaxf(v, 0.f);
        Out[(size_t)r * HD + c] = v;
      }
    }
  }
}

// ---------------- elementwise / scatter kernels ----------------
__global__ void zerok(float* __restrict__ p, int n) {
  int i = blockIdx.x * 256 + threadIdx.x;
  if (i < n) p[i] = 0.f;
}
__global__ void copyk(float* __restrict__ d, const float* __restrict__ s, int n) {
  int i = blockIdx.x * 256 + threadIdx.x;
  if (i < n) d[i] = s[i];
}

// h = (atom_emb[tok] + dist_emb[dist] + relu(lp[f/12]*logp_w + logp_b)) * valid
__global__ void encodek(const int* __restrict__ xt, const int* __restrict__ ds,
                        const int* __restrict__ nid, const float* __restrict__ lp,
                        const float* __restrict__ atom, const float* __restrict__ de,
                        const float* __restrict__ lw, const float* __restrict__ lb,
                        float* __restrict__ h)
{
  int idx = blockIdx.x * 256 + threadIdx.x;
  if (idx >= FF * HD) return;
  int f = idx >> 7, c = idx & 127;
  float valid = (nid[f] >= 0) ? 1.f : 0.f;
  float pe = fmaxf(lp[f / KSUB] * lw[c] + lb[c], 0.f);
  h[idx] = (atom[xt[f] * HD + c] + de[ds[f] * HD + c] + pe) * valid;
}

// GINE message pass: Outp[dst] += relu(X[src] + EA)   (Outp pre-initialized with X)
__global__ void edge_msg(const float* __restrict__ X, const int* __restrict__ src,
                         const int* __restrict__ dst, const float* __restrict__ EA,
                         float* __restrict__ Outp, int E)
{
  int idx = blockIdx.x * 256 + threadIdx.x;
  if (idx >= E * HD) return;
  int e = idx >> 7, c = idx & 127;
  float v = fmaxf(X[(size_t)src[e] * HD + c] + EA[(size_t)e * HD + c], 0.f);
  atomicAdd(&Outp[(size_t)dst[e] * HD + c], v);
}

// BatchNorm column stats: st[0:128]=sum, st[128:256]=sumsq
__global__ __launch_bounds__(256) void bn_stats(const float* __restrict__ X, int rows,
                                                float* __restrict__ st)
{
  int tid = threadIdx.x;
  int c = tid & 127;
  int r0 = blockIdx.x * 256 + (tid >> 7);
  float s = 0.f, sq = 0.f;
  for (int i = 0; i < 128; ++i) {
    int r = r0 + i * 2;
    if (r < rows) { float v = X[(size_t)r * HD + c]; s += v; sq += v * v; }
  }
  atomicAdd(&st[c], s);
  atomicAdd(&st[128 + c], sq);
}

__global__ void bn_fin(const float* __restrict__ st, int rows,
                       const float* __restrict__ g, const float* __restrict__ b,
                       float* __restrict__ sc, float* __restrict__ sh)
{
  int c = threadIdx.x;                 // 128 threads
  float inv = 1.f / (float)rows;
  float mu = st[c] * inv;
  float var = st[128 + c] * inv - mu * mu;
  float rs = rsqrtf(var + 1e-5f);
  float s = g[c] * rs;
  sc[c] = s;
  sh[c] = b[c] - mu * s;
}

// HT inter-root weight, pass 1: w_un + segment sum over canonical ids
__global__ void htw1k(const int* __restrict__ nid, const float* __restrict__ lp,
                      const float* __restrict__ a_inter, float* __restrict__ wun,
                      float* __restrict__ wsum)
{
  int s = blockIdx.x * 256 + threadIdx.x;
  if (s >= SSUB) return;
  int rootid = nid[s * KSUB];
  float wu = (rootid >= 0) ? __expf(-a_inter[0] * lp[s]) : 0.f;
  wun[s] = wu;
  if (rootid >= 0) atomicAdd(&wsum[rootid], wu);
}
__global__ void htw2k(const int* __restrict__ nid, const float* __restrict__ wun,
                      const float* __restrict__ wsum, float* __restrict__ htw)
{
  int s = blockIdx.x * 256 + threadIdx.x;
  if (s >= SSUB) return;
  int rootid = nid[s * KSUB];
  htw[s] = (rootid >= 0) ? wun[s] / (wsum[rootid] + 1e-16f) : 0.f;
}

// HT-weighted scatter of root reps to canonical nodes
__global__ void canonk(const float* __restrict__ h, const int* __restrict__ nid,
                       const float* __restrict__ htw, float* __restrict__ hc)
{
  int idx = blockIdx.x * 256 + threadIdx.x;
  if (idx >= SSUB * HD) return;
  int s = idx >> 7, c = idx & 127;
  float w = htw[s];
  if (w != 0.f) {
    int rootid = nid[s * KSUB];                     // >= 0 when w != 0
    atomicAdd(&hc[(size_t)rootid * HD + c], h[(size_t)(s * KSUB) * HD + c] * w);
  }
}

// Fuses both BN applies + root/non-root select + relu + valid mask
__global__ void combinek(const float* __restrict__ h1r, const float* __restrict__ hnr,
                         const float* __restrict__ hi, const int* __restrict__ nid,
                         const float* __restrict__ sc1, const float* __restrict__ sh1,
                         const float* __restrict__ sc2, const float* __restrict__ sh2,
                         float* __restrict__ h)
{
  int idx = blockIdx.x * 256 + threadIdx.x;
  if (idx >= FF * HD) return;
  int f = idx >> 7, c = idx & 127;
  int id = nid[f];
  float valid = (id >= 0) ? 1.f : 0.f;
  float h1 = (h1r[idx] * sc1[c] + sh1[c]) * valid;
  float o;
  if (f % KSUB == 0) {
    int cid = (id >= 0) ? id : 0;
    o = h1 + (hi[(size_t)cid * HD + c] * sc2[c] + sh2[c]) * valid;
  } else {
    o = h1 + hnr[idx];
  }
  h[idx] = fmaxf(o, 0.f) * valid;
}

// readout: per-subgraph sum over the 12 nodes
__global__ void hsubk(const float* __restrict__ h, float* __restrict__ hs)
{
  int idx = blockIdx.x * 256 + threadIdx.x;
  if (idx >= SSUB * HD) return;
  int s = idx >> 7, c = idx & 127;
  float a = 0.f;
#pragma unroll
  for (int k = 0; k < KSUB; ++k) a += h[(size_t)(s * KSUB + k) * HD + c];
  hs[idx] = a;
}

__global__ void poolwk(const float* __restrict__ lp, const float* __restrict__ a_pool,
                       float* __restrict__ wp)
{
  int n = blockIdx.x * 256 + threadIdx.x;
  if (n >= NTOT) return;
  float a = a_pool[0];
  float t[MSUB], mx = -1e30f;
#pragma unroll
  for (int m = 0; m < MSUB; ++m) { t[m] = -a * lp[n * MSUB + m]; mx = fmaxf(mx, t[m]); }
  float se = 0.f;
#pragma unroll
  for (int m = 0; m < MSUB; ++m) { t[m] = __expf(t[m] - mx); se += t[m]; }
  float inv = 1.f / se;
#pragma unroll
  for (int m = 0; m < MSUB; ++m) wp[n * MSUB + m] = t[m] * inv;
}

__global__ void nodeembk(const float* __restrict__ hs, const float* __restrict__ wp,
                         float* __restrict__ ne)
{
  int idx = blockIdx.x * 256 + threadIdx.x;
  if (idx >= NTOT * HD) return;
  int n = idx >> 7, c = idx & 127;
  float a = 0.f;
#pragma unroll
  for (int m = 0; m < MSUB; ++m)
    a += wp[n * MSUB + m] * hs[(size_t)(n * MSUB + m) * HD + c];
  ne[idx] = a;
}

__global__ void finalk(const float* __restrict__ ne, const float* __restrict__ sc,
                       const float* __restrict__ sh, float* __restrict__ out)
{
  int idx = blockIdx.x * 256 + threadIdx.x;
  if (idx >= NTOT * HD) return;
  int n = idx >> 7, c = idx & 127;
  float v = ne[idx] * sc[c] + sh[c];
  atomicAdd(&out[(size_t)(n / (NTOT / BB)) * HD + c], v);   // batch_ids = n/50
}

// =====================================================================
extern "C" void kernel_launch(void* const* d_in, const int* in_sizes, int n_in,
                              void* d_out, int out_size, void* d_ws, size_t ws_size,
                              hipStream_t stream)
{
  (void)in_sizes; (void)n_in; (void)out_size; (void)ws_size;
  // ----- inputs (setup_inputs order) -----
  const int*   x_tok     = (const int*)  d_in[0];
  const int*   dist      = (const int*)  d_in[1];
  const int*   node_ids  = (const int*)  d_in[2];
  // d_in[3] valid (bool)      -> recomputed from node_ids
  // d_in[4] sub_batch, [5] root_flat_idx, [8] batch_ids -> analytic (f/12, s*12, n/50)
  const int*   intra_ei  = (const int*)  d_in[6];   // [2,EINTRA] row-major
  const int*   edge_index= (const int*)  d_in[7];   // [2,EINTER]
  const float* lp        = (const float*)d_in[9];
  const float* ea_flat   = (const float*)d_in[10];
  const float* edge_attr = (const float*)d_in[11];
  const float* atom_emb  = (const float*)d_in[12];
  const float* dist_emb  = (const float*)d_in[13];
  const float* logp_w    = (const float*)d_in[14];
  const float* logp_b    = (const float*)d_in[15];
  const float* intra_W1  = (const float*)d_in[16];
  const float* intra_b1  = (const float*)d_in[17];
  const float* intra_W2  = (const float*)d_in[18];
  const float* intra_b2  = (const float*)d_in[19];
  const float* intra_bn_g= (const float*)d_in[20];
  const float* intra_bn_b= (const float*)d_in[21];
  const float* self_W    = (const float*)d_in[22];
  const float* self_b    = (const float*)d_in[23];
  const float* root_W    = (const float*)d_in[24];
  const float* root_b    = (const float*)d_in[25];
  const float* inter_W1  = (const float*)d_in[26];
  const float* inter_b1  = (const float*)d_in[27];
  const float* inter_W2  = (const float*)d_in[28];
  const float* inter_b2  = (const float*)d_in[29];
  const float* inter_bn_g= (const float*)d_in[30];
  const float* inter_bn_b= (const float*)d_in[31];
  const float* ro_bn_g   = (const float*)d_in[32];
  const float* ro_bn_b   = (const float*)d_in[33];
  const float* alpha_pool= (const float*)d_in[34];
  const float* alpha_int = (const float*)d_in[35];
  float* out = (float*)d_out;

  const int* e1src = intra_ei;              const int* e1dst = intra_ei + EINTRA;
  const int* e2src = edge_index;            const int* e2dst = edge_index + EINTER;

  // ----- workspace carve -----
  const size_t FH = (size_t)FF * HD, NH = (size_t)NTOT * HD, SH = (size_t)SSUB * HD;
  float* p    = (float*)d_ws;
  float* h    = p;            p += FH;
  float* buf1 = p;            p += FH;     // hin -> h1raw
  float* buf2 = p;            p += FH;     // t1  -> t2/h_non_root
  float* hc   = p;            p += NH;
  float* hin2 = p;            p += NH;
  float* ti1  = p;            p += NH;
  float* hi   = p;            p += NH;     // h_inter (pre-BN)
  float* hs   = p;            p += SH;
  float* ne   = p;            p += NH;
  float* wun  = p;            p += SSUB;
  float* htw  = p;            p += SSUB;
  float* wp   = p;            p += SSUB;
  float* wsum = p;            p += NTOT;
  float* stI  = p;            p += 512;    // intra BN: sum,sumsq,scale,shift
  float* stJ  = p;            p += 512;    // inter BN
  float* stR  = p;            p += 512;    // readout BN

  const int gFH = (int)((FH + 255) / 256);        // 60000
  const int gNH = (int)((NH + 255) / 256);        // 1250
  const int gSH = (int)((SH + 255) / 256);        // 5000
  const int gE1 = EINTRA * HD / 256;              // 120000
  const int gE2 = EINTER * HD / 256;              // 20000
  const int gGF = (FF + 127) / 128;               // 938
  const int gGN = (NTOT + 127) / 128;             // 20
  const int gS  = (SSUB + 255) / 256;
  const int gN  = (NTOT + 255) / 256;

  // ----- input encoding -----
  encodek<<<gFH, 256, 0, stream>>>(x_tok, dist, node_ids, lp, atom_emb, dist_emb,
                                   logp_w, logp_b, h);

  // ----- HT inter-root weights -----
  zerok<<<(NTOT + 255) / 256, 256, 0, stream>>>(wsum, NTOT);
  htw1k<<<gS, 256, 0, stream>>>(node_ids, lp, alpha_int, wun, wsum);
  htw2k<<<gS, 256, 0, stream>>>(node_ids, wun, wsum, htw);

  // ----- layers -----
  for (int l = 0; l < LL; ++l) {
    const size_t oW = (size_t)l * HD * HD, oB = (size_t)l * HD;

    // intra GINE: hin = h + sum relu(h[src]+ea)
    copyk<<<gFH, 256, 0, stream>>>(buf1, h, (int)FH);
    edge_msg<<<gE1, 256, 0, stream>>>(h, e1src, e1dst, ea_flat, buf1, EINTRA);
    // MLP: t1 = relu(hin@W1+b1); h1raw = t1@W2+b2
    gemm128_wmma<<<gGF, 256, 0, stream>>>(buf1, intra_W1 + oW, intra_b1 + oB,
                                          nullptr, buf2, FF, 0, 1);
    gemm128_wmma<<<gGF, 256, 0, stream>>>(buf2, intra_W2 + oW, intra_b2 + oB,
                                          nullptr, buf1, FF, 0, 0);
    // intra BN stats
    zerok<<<1, 256, 0, stream>>>(stI, 256);
    bn_stats<<<(FF + 255) / 256, 256, 0, stream>>>(buf1, FF, stI);
    bn_fin<<<1, 128, 0, stream>>>(stI, FF, intra_bn_g + oB, intra_bn_b + oB,
                                  stI + 256, stI + 384);
    // h_non_root = h@self_W+self_b + h[rootrow]@root_W+root_b
    gemm128_wmma<<<gGF, 256, 0, stream>>>(h, self_W + oW, self_b + oB,
                                          nullptr, buf2, FF, 0, 0);
    gemm128_wmma<<<gGF, 256, 0, stream>>>(h, root_W + oW, root_b + oB,
                                          buf2, buf2, FF, 1, 0);
    // canonical roots + inter GINE
    zerok<<<gNH, 256, 0, stream>>>(hc, (int)NH);
    canonk<<<gSH, 256, 0, stream>>>(h, node_ids, htw, hc);
    copyk<<<gNH, 256, 0, stream>>>(hin2, hc, (int)NH);
    edge_msg<<<gE2, 256, 0, stream>>>(hc, e2src, e2dst, edge_attr, hin2, EINTER);
    gemm128_wmma<<<gGN, 256, 0, stream>>>(hin2, inter_W1 + oW, inter_b1 + oB,
                                          nullptr, ti1, NTOT, 0, 1);
    gemm128_wmma<<<gGN, 256, 0, stream>>>(ti1, inter_W2 + oW, inter_b2 + oB,
                                          nullptr, hi, NTOT, 0, 0);
    zerok<<<1, 256, 0, stream>>>(stJ, 256);
    bn_stats<<<(NTOT + 255) / 256, 256, 0, stream>>>(hi, NTOT, stJ);
    bn_fin<<<1, 128, 0, stream>>>(stJ, NTOT, inter_bn_g + oB, inter_bn_b + oB,
                                  stJ + 256, stJ + 384);
    // fuse BN applies + select + relu + mask
    combinek<<<gFH, 256, 0, stream>>>(buf1, buf2, hi, node_ids,
                                      stI + 256, stI + 384, stJ + 256, stJ + 384, h);
  }

  // ----- readout -----
  hsubk<<<gSH, 256, 0, stream>>>(h, hs);
  poolwk<<<gN, 256, 0, stream>>>(lp, alpha_pool, wp);
  nodeembk<<<gNH, 256, 0, stream>>>(hs, wp, ne);
  zerok<<<1, 256, 0, stream>>>(stR, 256);
  bn_stats<<<(NTOT + 255) / 256, 256, 0, stream>>>(ne, NTOT, stR);
  bn_fin<<<1, 128, 0, stream>>>(stR, NTOT, ro_bn_g, ro_bn_b, stR + 256, stR + 384);
  zerok<<<(BB * HD + 255) / 256, 256, 0, stream>>>(out, BB * HD);
  finalk<<<gNH, 256, 0, stream>>>(ne, stR + 256, stR + 384, out);
}
